// MultiHeadedAttention_22686017257945
// MI455X (gfx1250) — matrix operand order repaired and checked by reading, not compiled, not run on hardware
//
#include <hip/hip_runtime.h>

// ---------------------------------------------------------------------------
// Problem constants (from reference setup_inputs)
// ---------------------------------------------------------------------------
#define B_   2
#define N_   3200
#define DIN  512
#define H_   8
#define D_   64
#define T_   128
#define J_   25
#define ROWS (B_ * N_)          // 6400 token rows for QKV GEMM
#define HD   (H_ * D_)          // 512 output cols

typedef __attribute__((ext_vector_type(4)))  unsigned int  u32x4;
typedef __attribute__((ext_vector_type(16))) __bf16        bf16x16;
typedef __attribute__((ext_vector_type(8)))  float         f32x8;

union FragAB {
  u32x4          u4[2];
  unsigned short h[16];
  bf16x16        bf;
};

__device__ __forceinline__ unsigned short f32_to_bf16(float f) {
  unsigned int u = __float_as_uint(f);
  u += 0x7FFFu + ((u >> 16) & 1u);     // round to nearest even
  return (unsigned short)(u >> 16);
}

// ---------------------------------------------------------------------------
// Kernel 1a: x f32 -> bf16
// ---------------------------------------------------------------------------
__global__ void cvt_x_kernel(const float* __restrict__ x,
                             unsigned short* __restrict__ xb, int n) {
  int i = blockIdx.x * blockDim.x + threadIdx.x;
  if (i < n) xb[i] = f32_to_bf16(x[i]);
}

// ---------------------------------------------------------------------------
// Kernel 1b: W [K,N] f32 -> WT [N,K] bf16 (transposed so B-fragments are
// contiguous 32B per lane)
// ---------------------------------------------------------------------------
__global__ void cvt_wt_kernel(const float* __restrict__ Wq,
                              const float* __restrict__ Wk,
                              const float* __restrict__ Wv,
                              unsigned short* __restrict__ wtq,
                              unsigned short* __restrict__ wtk,
                              unsigned short* __restrict__ wtv) {
  int kk = blockIdx.x * blockDim.x + threadIdx.x;   // K index (contiguous write)
  int nn = blockIdx.y * blockDim.y + threadIdx.y;   // N index
  const float* W = (blockIdx.z == 0) ? Wq : ((blockIdx.z == 1) ? Wk : Wv);
  unsigned short* WT = (blockIdx.z == 0) ? wtq : ((blockIdx.z == 1) ? wtk : wtv);
  WT[(size_t)nn * DIN + kk] = f32_to_bf16(W[(size_t)kk * DIN + nn]);
}

// ---------------------------------------------------------------------------
// Kernel 2: fused QKV projection, 32x64 register tile per wave.
//   [6400,512]bf16 x [512,512]bf16 (+bias, ReLU for V) -> bf16 [B,H,N,D]
// Per k-step: 2 A-frags + 4 B-frags (12 x b128) feed 8 WMMAs (load:wmma 1.5).
// blockIdx.z selects Q / K / V.  Block = 4 waves covering 128 rows x 64 cols.
// ---------------------------------------------------------------------------
__global__ void qkv_gemm_kernel(const unsigned short* __restrict__ xb,
                                const unsigned short* __restrict__ wtq,
                                const unsigned short* __restrict__ wtk,
                                const unsigned short* __restrict__ wtv,
                                const float* __restrict__ bq,
                                const float* __restrict__ bk,
                                const float* __restrict__ bv,
                                unsigned short* __restrict__ qb,
                                unsigned short* __restrict__ kb,
                                unsigned short* __restrict__ vb) {
  const int which = blockIdx.z;
  const unsigned short* wt  = (which == 0) ? wtq : ((which == 1) ? wtk : wtv);
  const float*          bia = (which == 0) ? bq  : ((which == 1) ? bk  : bv);
  unsigned short*       out = (which == 0) ? qb  : ((which == 1) ? kb  : vb);
  const bool relu = (which == 2);

  const int wave = threadIdx.x >> 5;
  const int lane = threadIdx.x & 31;
  const int half = lane >> 4;     // which 16-lane half of the wave
  const int l16  = lane & 15;

  const int mbase = blockIdx.x * 128 + wave * 32;   // 32 token rows per wave
  const int nbase = blockIdx.y * 64;                // 64 output cols per wave

  // A-fragment source rows (contiguous k order, matching B:
  //  lane<16 holds K k..k+15; lane>=16 K k+16..k+31)
  const unsigned short* arow0 = xb + (size_t)(mbase + l16) * DIN;
  const unsigned short* arow1 = arow0 + (size_t)16 * DIN;
  // B-fragment source columns (lane<16 holds K k..k+15; lane>=16 K k+16..k+31)
  const unsigned short* bcol0 = wt + (size_t)(nbase + l16) * DIN + half * 16;
  const unsigned short* bcol1 = bcol0 + (size_t)16 * DIN;
  const unsigned short* bcol2 = bcol0 + (size_t)32 * DIN;
  const unsigned short* bcol3 = bcol0 + (size_t)48 * DIN;

  f32x8 zf = {0.f, 0.f, 0.f, 0.f, 0.f, 0.f, 0.f, 0.f};
  f32x8 acc[2][4];
#pragma unroll
  for (int mi = 0; mi < 2; ++mi)
#pragma unroll
    for (int ni = 0; ni < 4; ++ni) acc[mi][ni] = zf;

#pragma unroll 2
  for (int k = 0; k < DIN; k += 32) {
    // prefetch two k-steps ahead (lowers to global_prefetch_b8)
    __builtin_prefetch(arow0 + k + 64, 0, 1);
    __builtin_prefetch(arow1 + k + 64, 0, 1);
    __builtin_prefetch(bcol0 + k + 64, 0, 1);
    __builtin_prefetch(bcol2 + k + 64, 0, 1);

    FragAB a0, a1, b0, b1, b2, b3;
    a0.u4[0] = *(const u32x4*)(arow0 + k + half * 16);
    a0.u4[1] = *(const u32x4*)(arow0 + k + 8 + half * 16);
    a1.u4[0] = *(const u32x4*)(arow1 + k + half * 16);
    a1.u4[1] = *(const u32x4*)(arow1 + k + 8 + half * 16);
    b0.u4[0] = *(const u32x4*)(bcol0 + k);
    b0.u4[1] = *(const u32x4*)(bcol0 + k + 8);
    b1.u4[0] = *(const u32x4*)(bcol1 + k);
    b1.u4[1] = *(const u32x4*)(bcol1 + k + 8);
    b2.u4[0] = *(const u32x4*)(bcol2 + k);
    b2.u4[1] = *(const u32x4*)(bcol2 + k + 8);
    b3.u4[0] = *(const u32x4*)(bcol3 + k);
    b3.u4[1] = *(const u32x4*)(bcol3 + k + 8);

    acc[0][0] = __builtin_amdgcn_wmma_f32_16x16x32_bf16(false, a0.bf, false, b0.bf, (short)0, acc[0][0], false, false);
    acc[0][1] = __builtin_amdgcn_wmma_f32_16x16x32_bf16(false, a0.bf, false, b1.bf, (short)0, acc[0][1], false, false);
    acc[0][2] = __builtin_amdgcn_wmma_f32_16x16x32_bf16(false, a0.bf, false, b2.bf, (short)0, acc[0][2], false, false);
    acc[0][3] = __builtin_amdgcn_wmma_f32_16x16x32_bf16(false, a0.bf, false, b3.bf, (short)0, acc[0][3], false, false);
    acc[1][0] = __builtin_amdgcn_wmma_f32_16x16x32_bf16(false, a1.bf, false, b0.bf, (short)0, acc[1][0], false, false);
    acc[1][1] = __builtin_amdgcn_wmma_f32_16x16x32_bf16(false, a1.bf, false, b1.bf, (short)0, acc[1][1], false, false);
    acc[1][2] = __builtin_amdgcn_wmma_f32_16x16x32_bf16(false, a1.bf, false, b2.bf, (short)0, acc[1][2], false, false);
    acc[1][3] = __builtin_amdgcn_wmma_f32_16x16x32_bf16(false, a1.bf, false, b3.bf, (short)0, acc[1][3], false, false);
  }

  // Epilogue. Whole 32-row wave tile is in one batch (N_ % 32 == 0): hoist.
  const int bb  = (mbase >= N_) ? 1 : 0;
  const int nn0 = mbase - bb * N_;
#pragma unroll
  for (int ni = 0; ni < 4; ++ni) {
    const int   gc  = nbase + ni * 16 + l16;   // global col in [0,512)
    const float bvv = bia[gc];
    const int   hh  = gc >> 6;                 // head
    const int   dd  = gc & 63;                 // dim within head
    unsigned short* op =
        out + ((size_t)(bb * H_ + hh) * N_ + nn0) * D_ + dd;
#pragma unroll
    for (int mi = 0; mi < 2; ++mi) {
#pragma unroll
      for (int r = 0; r < 8; ++r) {
        const int mrow = mi * 16 + r + half * 8;   // row within wave tile
        float v = acc[mi][ni][r] + bvv;
        if (relu) v = fmaxf(v, 0.0f);
        op[(size_t)mrow * D_] = f32_to_bf16(v);
      }
    }
  }
}

// ---------------------------------------------------------------------------
// Kernel 3: per-frame masked attention. One wave per (frame, b*H+h).
// 25x25 block padded to 32x32; scores + PV both via bf16 WMMA.
// ---------------------------------------------------------------------------
__global__ void attn_kernel(const unsigned short* __restrict__ qb,
                            const unsigned short* __restrict__ kb,
                            const unsigned short* __restrict__ vb,
                            float* __restrict__ out) {
  __shared__ __align__(16) float          s_sc[32 * 32];   // raw scores
  __shared__ __align__(16) unsigned short s_p[32 * 32];    // softmax probs bf16
  __shared__ __align__(16) unsigned short s_vt[D_ * 32];   // V^T: [d][m]

  const int frame = blockIdx.x;
  const int bh    = blockIdx.y;       // b*H + h
  const int bb    = bh >> 3;
  const int hh    = bh & 7;
  const int row0  = frame * J_;

  const int lane = threadIdx.x;
  const int half = lane >> 4;
  const int l16  = lane & 15;

  const unsigned short* Qp = qb + (size_t)bh * N_ * D_;
  const unsigned short* Kp = kb + (size_t)bh * N_ * D_;
  const unsigned short* Vp = vb + (size_t)bh * N_ * D_;

  // Stage V^T into LDS (pad rows >= J_ with zeros).
  {
    const int m = lane;
    if (m < J_) {
      const unsigned short* vr = Vp + (size_t)(row0 + m) * D_;
#pragma unroll 8
      for (int d = 0; d < D_; ++d) s_vt[d * 32 + m] = vr[d];
    } else {
#pragma unroll 8
      for (int d = 0; d < D_; ++d) s_vt[d * 32 + m] = 0;
    }
  }

  // ---- scores S = Q K^T (2x2 tiles of 16x16, K-dim = 64 in 2 steps) ----
  const u32x4 z4 = {0u, 0u, 0u, 0u};
  f32x8 zf = {0.f, 0.f, 0.f, 0.f, 0.f, 0.f, 0.f, 0.f};
  f32x8 acc[2][2];
  acc[0][0] = zf; acc[0][1] = zf; acc[1][0] = zf; acc[1][1] = zf;

#pragma unroll
  for (int k = 0; k < D_; k += 32) {
#pragma unroll
    for (int mi = 0; mi < 2; ++mi) {
      FragAB a;
      const int rowA = mi * 16 + l16;
      if (rowA < J_) {
        const unsigned short* qr = Qp + (size_t)(row0 + rowA) * D_ + k;
        a.u4[0] = *(const u32x4*)(qr + half * 16);
        a.u4[1] = *(const u32x4*)(qr + 8 + half * 16);
      } else {
        a.u4[0] = z4; a.u4[1] = z4;
      }
#pragma unroll
      for (int ni = 0; ni < 2; ++ni) {
        FragAB b;
        const int colB = ni * 16 + l16;   // key index (column of S)
        if (colB < J_) {
          const unsigned short* kr =
              Kp + (size_t)(row0 + colB) * D_ + k + half * 16;
          b.u4[0] = *(const u32x4*)(kr);
          b.u4[1] = *(const u32x4*)(kr + 8);
        } else {
          b.u4[0] = z4; b.u4[1] = z4;
        }
        acc[mi][ni] = __builtin_amdgcn_wmma_f32_16x16x32_bf16(
            false, a.bf, false, b.bf, (short)0, acc[mi][ni], false, false);
      }
    }
  }

  // Spill scores to LDS: element r of acc[mi][ni] -> (M = mi*16+r+half*8, N = ni*16+l16)
#pragma unroll
  for (int mi = 0; mi < 2; ++mi)
#pragma unroll
    for (int ni = 0; ni < 2; ++ni)
#pragma unroll
      for (int r = 0; r < 8; ++r)
        s_sc[(mi * 16 + r + half * 8) * 32 + ni * 16 + l16] = acc[mi][ni][r];

  __syncthreads();   // single wave: lowers to NOP, keeps LDS ordering explicit

  // ---- masked softmax (scale 1/sqrt(64); cols >= J_ are masked out) ----
  {
    const int m = lane;                 // each lane owns one row
    float mx = -3.4e38f;
    for (int c = 0; c < J_; ++c) mx = fmaxf(mx, s_sc[m * 32 + c]);
    float sum = 0.f;
    for (int c = 0; c < 32; ++c) {
      float pv = 0.f;
      if (c < J_) { pv = __expf((s_sc[m * 32 + c] - mx) * 0.125f); sum += pv; }
      s_sc[m * 32 + c] = pv;
    }
    const float inv = 1.0f / sum;
    for (int c = 0; c < 32; ++c)
      s_p[m * 32 + c] = f32_to_bf16(s_sc[m * 32 + c] * inv);
  }

  __syncthreads();

  // ---- out = P V  (2 row tiles x 4 d-tiles, K-dim = 32 padded keys) ----
#pragma unroll
  for (int mi = 0; mi < 2; ++mi) {
    FragAB a;
    const int rowA = mi * 16 + l16;
    a.u4[0] = *(const u32x4*)(&s_p[rowA * 32 + half * 16]);
    a.u4[1] = *(const u32x4*)(&s_p[rowA * 32 + 8 + half * 16]);
#pragma unroll
    for (int nd = 0; nd < 4; ++nd) {
      FragAB b;
      const unsigned short* vtp = &s_vt[(nd * 16 + l16) * 32 + half * 16];
      b.u4[0] = *(const u32x4*)(vtp);
      b.u4[1] = *(const u32x4*)(vtp + 8);
      f32x8 o = zf;
      o = __builtin_amdgcn_wmma_f32_16x16x32_bf16(false, a.bf, false, b.bf,
                                                  (short)0, o, false, false);
#pragma unroll
      for (int r = 0; r < 8; ++r) {
        const int m = mi * 16 + r + half * 8;
        if (m < J_) {
          const int tok = row0 + m;
          const int d   = nd * 16 + l16;
          out[((size_t)(bb * N_) + tok) * HD + hh * D_ + d] = o[r];
        }
      }
    }
  }
}

// ---------------------------------------------------------------------------
// Host-side launcher
// ---------------------------------------------------------------------------
extern "C" void kernel_launch(void* const* d_in, const int* in_sizes, int n_in,
                              void* d_out, int out_size, void* d_ws, size_t ws_size,
                              hipStream_t stream) {
  (void)in_sizes; (void)n_in; (void)out_size; (void)ws_size;

  const float* x  = (const float*)d_in[0];
  const float* Wq = (const float*)d_in[1];
  const float* bq = (const float*)d_in[2];
  const float* Wk = (const float*)d_in[3];
  const float* bk = (const float*)d_in[4];
  const float* Wv = (const float*)d_in[5];
  const float* bv = (const float*)d_in[6];
  float* out = (float*)d_out;

  // Workspace carve-out (256B aligned): ~27.8 MB total
  char* ws = (char*)d_ws;
  size_t off = 0;
  auto carve = [&](size_t bytes) {
    void* p = ws + off;
    off += (bytes + 255) & ~(size_t)255;
    return p;
  };
  unsigned short* xb  = (unsigned short*)carve((size_t)ROWS * DIN * 2);
  unsigned short* wtq = (unsigned short*)carve((size_t)DIN * DIN * 2);
  unsigned short* wtk = (unsigned short*)carve((size_t)DIN * DIN * 2);
  unsigned short* wtv = (unsigned short*)carve((size_t)DIN * DIN * 2);
  unsigned short* qb  = (unsigned short*)carve((size_t)B_ * H_ * N_ * D_ * 2);
  unsigned short* kb  = (unsigned short*)carve((size_t)B_ * H_ * N_ * D_ * 2);
  unsigned short* vb  = (unsigned short*)carve((size_t)B_ * H_ * N_ * D_ * 2);

  // 1) bf16 conversions
  {
    const int n = ROWS * DIN;
    cvt_x_kernel<<<(n + 255) / 256, 256, 0, stream>>>(x, xb, n);
  }
  {
    dim3 blk(32, 8);
    dim3 grd(DIN / 32, DIN / 8, 3);
    cvt_wt_kernel<<<grd, blk, 0, stream>>>(Wq, Wk, Wv, wtq, wtk, wtv);
  }

  // 2) fused QKV projection: grid = (row blocks of 128, col blocks of 64, QKV)
  {
    dim3 blk(128);                       // 4 waves, one 32x64 tile each
    dim3 grd(ROWS / 128, HD / 64, 3);
    qkv_gemm_kernel<<<grd, blk, 0, stream>>>(xb, wtq, wtk, wtv,
                                             bq, bk, bv, qb, kb, vb);
  }

  // 3) per-frame masked attention: one wave per (frame, b*H+h)
  {
    dim3 blk(32);
    dim3 grd(T_, B_ * H_);
    attn_kernel<<<grd, blk, 0, stream>>>(qb, kb, vb, out);
  }
}
